// HeteroGAT_30133490549160
// MI455X (gfx1250) — matrix operand order repaired and checked by reading, not compiled, not run on hardware
//
#include <hip/hip_runtime.h>
#include <math.h>

typedef __attribute__((ext_vector_type(16))) _Float16 v16h;
typedef __attribute__((ext_vector_type(8)))  float    v8f;
typedef __attribute__((ext_vector_type(4)))  float    v4f;

#define IN_DIM  128
#define OUT_DIM 128

// ---------------------------------------------------------------- utilities

__device__ __forceinline__ unsigned enc_ord(float f) {
  // order-preserving float -> uint encoding (for atomic max on floats)
  unsigned u = __float_as_uint(f);
  return (u & 0x80000000u) ? ~u : (u | 0x80000000u);
}
__device__ __forceinline__ float dec_ord(unsigned e) {
  unsigned u = (e & 0x80000000u) ? (e & 0x7FFFFFFFu) : ~e;
  return __uint_as_float(u);
}
__device__ __forceinline__ float lrelu(float x) { return x > 0.f ? x : 0.01f * x; }

__global__ __launch_bounds__(256) void zero_f32(float* p, int n4) {
  int i = blockIdx.x * 256 + threadIdx.x;
  if (i < n4) {
    v4f z = {0.f, 0.f, 0.f, 0.f};
    ((v4f*)p)[i] = z;
  }
}

__global__ __launch_bounds__(256) void init_md(unsigned* mmax, float* denom, int n) {
  int i = blockIdx.x * 256 + threadIdx.x;
  if (i < n) { mmax[i] = 0x007FFFFFu; /* enc(-inf) */ denom[i] = 0.f; }
}

// Pre-swizzle W (f32 row-major [K=128][N=128]) into the WMMA B-fragment lane
// layout as f16: frag(t,c) covers N = t*16..t*16+15, K = c*32..c*32+31.
// Per 32-K chunk: lanes 0-15 hold K+0..15 (slot order), lanes 16-31 hold K+16..31.
__global__ __launch_bounds__(256) void pack_w(const float* __restrict__ W,
                                              _Float16* __restrict__ Bp) {
  int tid = blockIdx.x * 256 + threadIdx.x;     // 4096 threads total
  if (tid >= 128 * 32) return;
  int k  = tid >> 5;                            // 0..127
  int n4 = (tid & 31) * 4;                      // 0,4,...,124
  v4f w = *(const v4f*)(W + k * OUT_DIM + n4);
  int c  = k >> 5;
  int kr = k & 31;
  int hi = (kr >= 16) ? 1 : 0;
  int j  = kr & 15;                             // half-slot within lane
#pragma unroll
  for (int i = 0; i < 4; ++i) {
    int nn   = n4 + i;
    int t    = nn >> 4;
    int lane = (nn & 15) + hi * 16;
    Bp[(((t * 4 + c) * 32) + lane) * 16 + j] = (_Float16)w[i];
  }
}

// ---------------------------------------------------------------- GEMM (WMMA)
// wave = one 16-row M strip; 8 waves/block = 128 rows/block.
// Computes wh = x@W + b and fused scores s_src = wh·a[:128], s_dst = wh·a[128:].
__global__ __launch_bounds__(256) void gat_gemm(
    const float* __restrict__ x, const _Float16* __restrict__ Bp,
    const float* __restrict__ bias, const float* __restrict__ att,
    float* __restrict__ wh, float* __restrict__ s_src, float* __restrict__ s_dst,
    int n) {
  const int wave = threadIdx.x >> 5;
  const int lane = threadIdx.x & 31;
  const int row0 = (blockIdx.x * 8 + wave) * 16;
  if (row0 >= n) return;                        // uniform per wave
  const int l16 = lane & 15;
  const int hi  = lane >> 4;
  const int arow = row0 + l16;

  // A fragments for all 4 K-chunks (contiguous K order, matching pack_w):
  // lane<16: K = base+0..15 (elems 0-15); lane>=16: K = base+16..31.
  v16h A[4];
  const float* xr = x + (long)arow * IN_DIM;
#pragma unroll
  for (int c = 0; c < 4; ++c) {
    const float* p = xr + c * 32 + hi * 16;
    v4f f0 = *(const v4f*)(p);
    v4f f1 = *(const v4f*)(p + 4);
    v4f f2 = *(const v4f*)(p + 8);
    v4f f3 = *(const v4f*)(p + 12);
    v16h a;
#pragma unroll
    for (int i = 0; i < 4; ++i) {
      a[i]      = (_Float16)f0[i];
      a[4 + i]  = (_Float16)f1[i];
      a[8 + i]  = (_Float16)f2[i];
      a[12 + i] = (_Float16)f3[i];
    }
    A[c] = a;
  }

  float ssum[8], dsum[8];
#pragma unroll
  for (int r = 0; r < 8; ++r) { ssum[r] = 0.f; dsum[r] = 0.f; }

  const v16h* bp = (const v16h*)Bp;
#pragma unroll
  for (int t = 0; t < 8; ++t) {
    const int col = t * 16 + l16;
    const float bv = bias[col];
    v8f C = {bv, bv, bv, bv, bv, bv, bv, bv};
#pragma unroll
    for (int c = 0; c < 4; ++c) {
      v16h B = bp[(t * 4 + c) * 32 + lane];
      C = __builtin_amdgcn_wmma_f32_16x16x32_f16(false, A[c], false, B,
                                                 (short)0, C, false, false);
    }
    const float a0 = att[col];
    const float a1 = att[OUT_DIM + col];
#pragma unroll
    for (int r = 0; r < 8; ++r) {
      // C layout: VGPR r, lanes 0-15 -> M=r, lanes 16-31 -> M=8+r; N = l16
      wh[(long)(row0 + hi * 8 + r) * OUT_DIM + col] = C[r];
      ssum[r] += C[r] * a0;
      dsum[r] += C[r] * a1;
    }
  }
  // reduce scores across the 16 lanes of each half (xor masks < 16 stay in half)
#pragma unroll
  for (int r = 0; r < 8; ++r) {
    float v0 = ssum[r], v1 = dsum[r];
#pragma unroll
    for (int m = 8; m >= 1; m >>= 1) {
      v0 += __shfl_xor(v0, m, 32);
      v1 += __shfl_xor(v1, m, 32);
    }
    if (l16 == 0) {
      const int rr = row0 + hi * 8 + r;
      s_src[rr] = v0;
      s_dst[rr] = v1;
    }
  }
}

// ---------------------------------------------------------------- edge passes

__global__ __launch_bounds__(256) void edge_max(
    const int* __restrict__ src, const int* __restrict__ dst,
    const float* __restrict__ ss, const float* __restrict__ sd,
    unsigned* __restrict__ mmax, int e) {
  int i = blockIdx.x * 256 + threadIdx.x;
  if (i >= e) return;
  int d = dst[i];
  float v = lrelu(ss[src[i]] + sd[d]);
  atomicMax(&mmax[d], enc_ord(v));
}

__global__ __launch_bounds__(256) void edge_exp(
    const int* __restrict__ src, const int* __restrict__ dst,
    const float* __restrict__ ss, const float* __restrict__ sd,
    const unsigned* __restrict__ mmax, float* __restrict__ denom,
    float* __restrict__ exbuf, int e) {
  int i = blockIdx.x * 256 + threadIdx.x;
  if (i >= e) return;
  int d = dst[i];
  float v  = lrelu(ss[src[i]] + sd[d]);
  float ex = __expf(v - dec_ord(mmax[d]));
  exbuf[i] = ex;
  __hip_atomic_fetch_add(&denom[d], ex, __ATOMIC_RELAXED, __HIP_MEMORY_SCOPE_AGENT);
}

// one wave per edge: each lane handles 4 of the 128 features
__global__ __launch_bounds__(256) void edge_scatter(
    const int* __restrict__ src, const int* __restrict__ dst,
    const float* __restrict__ wh, const float* __restrict__ exbuf,
    const float* __restrict__ denom, float* __restrict__ out, int e) {
  int widx = (int)((blockIdx.x * 256 + threadIdx.x) >> 5);
  int lane = threadIdx.x & 31;
  if (widx >= e) return;
  int s = src[widx], d = dst[widx];
  float alpha = exbuf[widx] / denom[d];
  v4f w = *(const v4f*)(wh + (long)s * OUT_DIM + lane * 4);
  float* o = out + (long)d * OUT_DIM + lane * 4;
#pragma unroll
  for (int i = 0; i < 4; ++i)
    __hip_atomic_fetch_add(o + i, alpha * w[i], __ATOMIC_RELAXED,
                           __HIP_MEMORY_SCOPE_AGENT);
}

// ---------------------------------------------------------------- launch

extern "C" void kernel_launch(void* const* d_in, const int* in_sizes, int n_in,
                              void* d_out, int out_size, void* d_ws, size_t ws_size,
                              hipStream_t stream) {
  const float* x   = (const float*)d_in[0];
  const int* src0  = (const int*)d_in[1];
  const int* dst0  = (const int*)d_in[2];
  const int* src1  = (const int*)d_in[3];
  const int* dst1  = (const int*)d_in[4];
  const float* W0  = (const float*)d_in[5];
  const float* b0  = (const float*)d_in[6];
  const float* a0  = (const float*)d_in[7];
  const float* W1  = (const float*)d_in[8];
  const float* b1  = (const float*)d_in[9];
  const float* a1  = (const float*)d_in[10];

  const int N = in_sizes[0] / IN_DIM;   // 100000
  const int E = in_sizes[1];            // 320000
  float* out = (float*)d_out;

  char* ws = (char*)d_ws;
  size_t off = 0;
  auto alloc = [&](size_t bytes) -> char* {
    char* p = ws + off;
    off = (off + bytes + 255) & ~(size_t)255;
    return p;
  };
  float*    wh    = (float*)alloc((size_t)N * OUT_DIM * 4);
  float*    s_src = (float*)alloc((size_t)N * 4);
  float*    s_dst = (float*)alloc((size_t)N * 4);
  unsigned* mmax  = (unsigned*)alloc((size_t)N * 4);
  float*    denom = (float*)alloc((size_t)N * 4);
  float*    exbuf = (float*)alloc((size_t)E * 4);
  _Float16* Bp    = (_Float16*)alloc((size_t)IN_DIM * OUT_DIM * 2);

  const int n4 = N * OUT_DIM / 4;
  zero_f32<<<(n4 + 255) / 256, 256, 0, stream>>>(out, n4);

  const int*   srcs[2] = {src0, src1};
  const int*   dsts[2] = {dst0, dst1};
  const float* Ws[2]   = {W0, W1};
  const float* bs[2]   = {b0, b1};
  const float* as[2]   = {a0, a1};

  for (int r = 0; r < 2; ++r) {
    pack_w<<<16, 256, 0, stream>>>(Ws[r], Bp);
    init_md<<<(N + 255) / 256, 256, 0, stream>>>(mmax, denom, N);
    gat_gemm<<<(N + 127) / 128, 256, 0, stream>>>(x, Bp, bs[r], as[r], wh,
                                                  s_src, s_dst, N);
    edge_max<<<(E + 255) / 256, 256, 0, stream>>>(srcs[r], dsts[r], s_src, s_dst,
                                                  mmax, E);
    edge_exp<<<(E + 255) / 256, 256, 0, stream>>>(srcs[r], dsts[r], s_src, s_dst,
                                                  mmax, denom, exbuf, E);
    edge_scatter<<<(E + 7) / 8, 256, 0, stream>>>(srcs[r], dsts[r], wh, exbuf,
                                                  denom, out, E);
  }
}
